// HeadGroupedFusionGate_31877247271513
// MI455X (gfx1250) — matrix operand order repaired and checked by reading, not compiled, not run on hardware
//
#include <hip/hip_runtime.h>
#include <math.h>

// Problem constants (match reference)
#define BT    2
#define LT    2048
#define HH    16
#define DD    64
#define HIDW  1024
#define NBR   4
#define NTOK  (BT*LT)      // 4096 tokens
#define TWO_D 128          // MLP hidden width
#define TPB   4            // tokens per block in gate kernel

typedef __attribute__((ext_vector_type(16))) __bf16 v16bf;
typedef __attribute__((ext_vector_type(8)))  float  v8f;
typedef __attribute__((ext_vector_type(4)))  unsigned u32x4;
typedef __attribute__((ext_vector_type(8)))  int      i32x8;
typedef __attribute__((ext_vector_type(4)))  int      i32x4;

union ABregs { v16bf v; unsigned u[8]; };

__device__ __forceinline__ unsigned f2bf(float f) {
  unsigned u = __float_as_uint(f);
  return (u + 0x7FFFu + ((u >> 16) & 1u)) >> 16;   // round-to-nearest-even
}
__device__ __forceinline__ unsigned pack2(float lo, float hi) {
  return f2bf(lo) | (f2bf(hi) << 16);
}

// TDM: async-load a 2-D tile (rows x row_elems f32, fully contiguous) into LDS.
// D# built per CDNA5 ISA (group0 + group1; groups 2/3 zero).
__device__ __forceinline__ void tdm_load_2d(unsigned lds_off, const void* gptr,
                                            unsigned rows, unsigned row_elems) {
  unsigned long long ga = (unsigned long long)gptr;
  u32x4 g0;
  g0[0] = 1u;                                             // count=1, user mode
  g0[1] = lds_off;                                        // lds_addr (bytes)
  g0[2] = (unsigned)(ga & 0xFFFFFFFFu);                   // global_addr lo
  g0[3] = (unsigned)((ga >> 32) & 0x01FFFFFFu) | (2u << 30); // addr hi | type=2
  i32x8 g1;
  g1[0] = 0x20000;                    // workgroup_mask=0, data_size=2 (4 bytes)
  g1[1] = (int)(row_elems << 16);     // tensor_dim0[15:0]
  g1[2] = (int)(rows << 16);          // tensor_dim0 hi=0 | tensor_dim1[15:0]
  g1[3] = (int)(row_elems << 16);     // tensor_dim1 hi=0 | tile_dim0
  g1[4] = (int)rows;                  // tile_dim1 | tile_dim2=0
  g1[5] = (int)row_elems;             // tensor_dim0_stride lo32
  g1[6] = 0;                          // stride0 hi | stride1 lo
  g1[7] = 0;
  i32x4 z4 = {0, 0, 0, 0};
#if defined(__clang_major__) && (__clang_major__ >= 23)
  i32x8 z8 = {0, 0, 0, 0, 0, 0, 0, 0};
  __builtin_amdgcn_tensor_load_to_lds(g0, g1, z4, z4, z8, 0);
#else
  __builtin_amdgcn_tensor_load_to_lds(g0, g1, z4, z4, 0);
#endif
}

// ---------------- prep: swizzle W1 into WMMA B-operand layout ----------------
// B-operand per (kstep, half, n): 8 contiguous dwords, dword i = bf16 pair of
// rows (k0, k0+1), k0 = kbase + half*8 + 2*(i&3) + (i>>2)*16, at column n.
__global__ __launch_bounds__(256) void prep_kernel(
    const float* __restrict__ W1, unsigned* __restrict__ w1h_sw,
    unsigned* __restrict__ w1b_sw, float* __restrict__ w1s) {
  int idx = blockIdx.x * 256 + threadIdx.x;
  if (idx < 65536) {                       // hidden-part swizzle (rows 0..1023)
    int i = idx & 7, n = (idx >> 3) & 127, half = (idx >> 10) & 1, ks = idx >> 11;
    int k0 = ks * 32 + half * 8 + 2 * (i & 3) + (i >> 2) * 16;
    w1h_sw[idx] = pack2(W1[k0 * TWO_D + n], W1[(k0 + 1) * TWO_D + n]);
  } else if (idx < 65536 + 16384) {        // branch-part swizzle (rows 1792..2047)
    int o = idx - 65536;
    int i = o & 7, n = (o >> 3) & 127, half = (o >> 10) & 1, ks = o >> 11;
    int k0 = 1792 + ks * 32 + half * 8 + 2 * (i & 3) + (i >> 2) * 16;
    w1b_sw[o] = pack2(W1[k0 * TWO_D + n], W1[(k0 + 1) * TWO_D + n]);
  } else if (idx < 65536 + 16384 + 12 * TWO_D) {
    int j = idx - (65536 + 16384);
    int s = j >> 7, n = j & 127;
    float acc = 0.f;
    int base = 1024 + s * 64;
    for (int r = 0; r < 64; ++r) acc += W1[(base + r) * TWO_D + n];
    w1s[s * TWO_D + n] = acc;
  }
}

// ---------------- shared GEMM: hidden[4096,1024] @ W1h[1024,128] + b1 --------
// One wave per 16-row M-tile; A converted once into registers and reused
// across all 8 N-tiles (acc = 64 VGPRs). No LDS, no barriers.
__global__ __launch_bounds__(256) void gemm_hidden_kernel(
    const float* __restrict__ hidden, const unsigned* __restrict__ w1h_sw,
    const float* __restrict__ b1, float* __restrict__ shared_out) {
  const int t = threadIdx.x;
  const int wave = t >> 5, lane = t & 31;
  const int half = lane >> 4, mrow = lane & 15, l15 = lane & 15;
  const int M0 = (blockIdx.x * 8 + wave) * 16;
  const float* rowp = hidden + (size_t)(M0 + mrow) * HIDW;

  v8f acc[8];
#pragma unroll
  for (int j = 0; j < 8; ++j) acc[j] = (v8f){0.f,0.f,0.f,0.f,0.f,0.f,0.f,0.f};

  for (int ks = 0; ks < 32; ++ks) {
    const int kb = ks * 32;
    const float* ap = rowp + kb + half * 8;   // k = koff..koff+7, +16..+23
    float4 f0 = *(const float4*)(ap);
    float4 f1 = *(const float4*)(ap + 4);
    float4 f2 = *(const float4*)(ap + 16);
    float4 f3 = *(const float4*)(ap + 20);
    ABregs A;
    A.u[0] = pack2(f0.x, f0.y); A.u[1] = pack2(f0.z, f0.w);
    A.u[2] = pack2(f1.x, f1.y); A.u[3] = pack2(f1.z, f1.w);
    A.u[4] = pack2(f2.x, f2.y); A.u[5] = pack2(f2.z, f2.w);
    A.u[6] = pack2(f3.x, f3.y); A.u[7] = pack2(f3.z, f3.w);
#pragma unroll
    for (int j = 0; j < 8; ++j) {
      ABregs Bm;
      const uint4* bp =
        (const uint4*)&w1h_sw[(size_t)(((ks * 2 + half) * 128) + j * 16 + l15) * 8];
      uint4 b0 = bp[0], b1v = bp[1];
      Bm.u[0]=b0.x; Bm.u[1]=b0.y; Bm.u[2]=b0.z; Bm.u[3]=b0.w;
      Bm.u[4]=b1v.x; Bm.u[5]=b1v.y; Bm.u[6]=b1v.z; Bm.u[7]=b1v.w;
      acc[j] = __builtin_amdgcn_wmma_f32_16x16x32_bf16(
          false, A.v, false, Bm.v, (short)0, acc[j], false, false);
    }
  }
#pragma unroll
  for (int j = 0; j < 8; ++j) {
    const int n = j * 16 + l15;
    const float bb = b1[n];
#pragma unroll
    for (int r = 0; r < 8; ++r)
      shared_out[(M0 + r + half * 8) * TWO_D + n] = acc[j][r] + bb;
  }
}

// ---------------- fused gate kernel: TPB tokens per block -------------------
// TDM double-buffers the 4x16x64 f32 branch tiles into LDS; B operand lives
// in registers for the whole block.  Dynamic LDS layout (base offset 0):
//   [0]      btile : 2 x 4 x 16 x 64 f32  (32768 B)
//   [32768]  pk    : 16 x 128 dwords      ( 8192 B)  packed bf16 pairs
//   [40960]  stat  : 16 x 12 f32          (  768 B)
//   [41728]  plog  : 8 x 16 x 4 f32       ( 2048 B)
//   [43776]  logit : 16 x 4 f32           (  256 B)   -> total 44032 B
#define SM_BT   0
#define SM_PK   32768
#define SM_ST   40960
#define SM_PL   41728
#define SM_LG   43776
#define SM_TOT  44032

__global__ __launch_bounds__(256) void gate_kernel(
    const float* __restrict__ br0, const float* __restrict__ br1,
    const float* __restrict__ br2, const float* __restrict__ br3,
    const unsigned* __restrict__ w1b_sw, const float* __restrict__ w1s,
    const float* __restrict__ shared_in, const float* __restrict__ W2,
    const float* __restrict__ b2, const float* __restrict__ eps_floor,
    const float* __restrict__ temp, float* __restrict__ out) {
  extern __shared__ char smem[];
  float    (*btile)[NBR][HH][DD] = (float (*)[NBR][HH][DD])(smem + SM_BT);
  unsigned (*pk)[128]            = (unsigned (*)[128])(smem + SM_PK);
  float    (*stat)[12]           = (float (*)[12])(smem + SM_ST);
  float    (*plog)[HH][NBR]      = (float (*)[HH][NBR])(smem + SM_PL);
  float    (*logit)[NBR]         = (float (*)[NBR])(smem + SM_LG);

  const int t = threadIdx.x;
  const int wave = t >> 5, lane = t & 31;
  const int half = lane >> 4, mrow = lane & 15;
  const int n = wave * 16 + (lane & 15);
  const int tok0 = blockIdx.x * TPB;
  const float* brs[NBR] = {br0, br1, br2, br3};

  // Prologue: wave 0 DMAs token tok0's branch tiles into buffer 0.
  if (wave == 0) {
#pragma unroll
    for (int i = 0; i < NBR; ++i)
      tdm_load_2d(SM_BT + i * 4096, brs[i] + (size_t)tok0 * (HH * DD), HH, DD);
    __builtin_amdgcn_s_wait_tensorcnt(0);
  }

  // Block-invariant register state: B operand (64 VGPRs) + epilogue constants.
  uint4 Brg[8][2];
#pragma unroll
  for (int ks = 0; ks < 8; ++ks) {
    const uint4* bp =
      (const uint4*)&w1b_sw[(size_t)(((ks * 2 + half) * 128) + n) * 8];
    Brg[ks][0] = bp[0]; Brg[ks][1] = bp[1];
  }
  float ws12[12];
#pragma unroll
  for (int j = 0; j < 12; ++j) ws12[j] = w1s[j * TWO_D + n];
  float w2v[NBR];
#pragma unroll
  for (int q = 0; q < NBR; ++q) w2v[q] = W2[n * NBR + q];

  __syncthreads();                       // buffer 0 ready for everyone

  for (int tau = 0; tau < TPB; ++tau) {
    const int tok = tok0 + tau;
    const int cur = tau & 1;

    // Kick off DMA of the next token's tiles into the other buffer.
    if (wave == 0 && tau + 1 < TPB) {
#pragma unroll
      for (int i = 0; i < NBR; ++i)
        tdm_load_2d(SM_BT + (cur ^ 1) * 16384 + i * 4096,
                    brs[i] + (size_t)(tok + 1) * (HH * DD), HH, DD);
    }

    // Pack f32 -> bf16 pairs (each element converted exactly once).
#pragma unroll
    for (int q = 0; q < 8; ++q) {
      int e = q * 256 + t;
      int row = e >> 7, kp = e & 127;
      int k = 2 * kp, i = k >> 6, c = k & 63;
      pk[row][kp] = pack2(btile[cur][i][row][c], btile[cur][i][row][c + 1]);
    }
    // Per-(head, branch) stats in f32 (rotated start to dodge bank conflicts).
    if (t < 64) {
      const int r = t >> 2, i = t & 3;
      float s = 0.f, sq = 0.f, mx = -3.402823466e38f;
      for (int c0 = 0; c0 < DD; ++c0) {
        int c = (c0 + t) & (DD - 1);
        float x = btile[cur][i][r][c];
        s += x; sq += x * x; mx = fmaxf(mx, x);
      }
      stat[r][i * 3 + 0] = s * (1.f / 64.f);
      stat[r][i * 3 + 1] = sqrtf(fmaxf(sq * (1.f / 64.f), 1e-8f));
      stat[r][i * 3 + 2] = mx;
    }
    __syncthreads();

    // Branch GEMM: 16 heads x K=256 x N=128; B from registers.
    v8f acc = {0.f,0.f,0.f,0.f,0.f,0.f,0.f,0.f};
#pragma unroll
    for (int ks = 0; ks < 8; ++ks) {
      ABregs A, Bm;
      const uint4* ap = (const uint4*)&pk[mrow][ks * 16];
      uint4 a0 = ap[half], a1 = ap[2 + half];
      A.u[0]=a0.x; A.u[1]=a0.y; A.u[2]=a0.z; A.u[3]=a0.w;
      A.u[4]=a1.x; A.u[5]=a1.y; A.u[6]=a1.z; A.u[7]=a1.w;
      Bm.u[0]=Brg[ks][0].x; Bm.u[1]=Brg[ks][0].y;
      Bm.u[2]=Brg[ks][0].z; Bm.u[3]=Brg[ks][0].w;
      Bm.u[4]=Brg[ks][1].x; Bm.u[5]=Brg[ks][1].y;
      Bm.u[6]=Brg[ks][1].z; Bm.u[7]=Brg[ks][1].w;
      acc = __builtin_amdgcn_wmma_f32_16x16x32_bf16(
          false, A.v, false, Bm.v, (short)0, acc, false, false);
    }

    // Epilogue: + shared hidden term (incl. b1) + rank-12 stats, exact GELU.
    const float sh = shared_in[tok * TWO_D + n];
    float h1v[8];
#pragma unroll
    for (int r = 0; r < 8; ++r) {
      const int m = r + half * 8;          // head index of this C element
      float v = acc[r] + sh;
#pragma unroll
      for (int j = 0; j < 12; ++j) v += stat[m][j] * ws12[j];
      h1v[r] = 0.5f * v * (1.f + erff(v * 0.70710678118654752f));
    }

    // logits = h1 @ W2: shuffle-reduce within each 16-lane half.
#pragma unroll
    for (int r = 0; r < 8; ++r) {
      const int m = r + half * 8;
#pragma unroll
      for (int q = 0; q < NBR; ++q) {
        float p = h1v[r] * w2v[q];
        p += __shfl_xor(p, 1, 32);
        p += __shfl_xor(p, 2, 32);
        p += __shfl_xor(p, 4, 32);
        p += __shfl_xor(p, 8, 32);
        if ((lane & 15) == 0) plog[wave][m][q] = p;
      }
    }
    __syncthreads();

    if (t < 64) {
      const int m = t >> 2, q = t & 3;
      float s = b2[q];
#pragma unroll
      for (int w = 0; w < 8; ++w) s += plog[w][m][q];
      logit[m][q] = s;
    }
    __syncthreads();

    // Softmax / temperature / eps-floor / renorm, one thread per head.
    if (t < HH) {
      const int h = t;
      float tt = fminf(fmaxf(temp[h], 0.2f), 10.f);
      float a0 = logit[h][0] / tt, a1 = logit[h][1] / tt;
      float a2 = logit[h][2] / tt, a3 = logit[h][3] / tt;
      float mx = fmaxf(fmaxf(a0, a1), fmaxf(a2, a3));
      float e0 = expf(a0 - mx), e1 = expf(a1 - mx);
      float e2 = expf(a2 - mx), e3 = expf(a3 - mx);
      float inv = 1.f / (e0 + e1 + e2 + e3);
      float g0 = e0 * inv, g1 = e1 * inv, g2 = e2 * inv, g3 = e3 * inv;
      g0 = fmaxf(g0, fminf(fmaxf(eps_floor[h * 4 + 0], 1e-7f), 0.1f));
      g1 = fmaxf(g1, fminf(fmaxf(eps_floor[h * 4 + 1], 1e-7f), 0.1f));
      g2 = fmaxf(g2, fminf(fmaxf(eps_floor[h * 4 + 2], 1e-7f), 0.1f));
      g3 = fmaxf(g3, fminf(fmaxf(eps_floor[h * 4 + 3], 1e-7f), 0.1f));
      inv = 1.f / (g0 + g1 + g2 + g3);
      float* o = out + (size_t)(tok * HH + h) * 4;
      o[0] = g0 * inv; o[1] = g1 * inv; o[2] = g2 * inv; o[3] = g3 * inv;
    }

    // Drain next buffer's DMA; barrier also fences plog/logit/pk reuse.
    if (wave == 0 && tau + 1 < TPB) __builtin_amdgcn_s_wait_tensorcnt(0);
    __syncthreads();
  }
}

extern "C" void kernel_launch(void* const* d_in, const int* in_sizes, int n_in,
                              void* d_out, int out_size, void* d_ws, size_t ws_size,
                              hipStream_t stream) {
  const float* hidden    = (const float*)d_in[0];
  const float* br0       = (const float*)d_in[1];
  const float* br1       = (const float*)d_in[2];
  const float* br2       = (const float*)d_in[3];
  const float* br3       = (const float*)d_in[4];
  const float* W1        = (const float*)d_in[5];
  const float* b1        = (const float*)d_in[6];
  const float* W2        = (const float*)d_in[7];
  const float* b2        = (const float*)d_in[8];
  const float* eps_floor = (const float*)d_in[9];
  const float* temp      = (const float*)d_in[10];
  float* out = (float*)d_out;

  // Workspace layout (~2.43 MB total)
  char* ws = (char*)d_ws;
  float*    shared_ws = (float*)ws;                                     // 4096*128 f32
  unsigned* w1h_sw = (unsigned*)(ws + (size_t)NTOK * TWO_D * 4);        // 65536 u32
  unsigned* w1b_sw = (unsigned*)((char*)w1h_sw + (size_t)65536 * 4);    // 16384 u32
  float*    w1s    = (float*)((char*)w1b_sw + (size_t)16384 * 4);       // 12*128 f32

  prep_kernel<<<326, 256, 0, stream>>>(W1, w1h_sw, w1b_sw, w1s);
  gemm_hidden_kernel<<<(NTOK / 16) / 8, 256, 0, stream>>>(hidden, w1h_sw, b1, shared_ws);
  gate_kernel<<<NTOK / TPB, 256, SM_TOT, stream>>>(br0, br1, br2, br3, w1b_sw, w1s,
                                                   shared_ws, W2, b2, eps_floor,
                                                   temp, out);
}
